// MultiLinearCentroids_27462020891405
// MI455X (gfx1250) — matrix operand order repaired and checked by reading, not compiled, not run on hardware
//
#include <hip/hip_runtime.h>

// Fixed shapes from the reference harness
#define BB 512
#define CC 100
#define EE 2048
#define FF 128

// phase-2 tiling
#define KC 32            // K-chunk staged per iteration
#define PITCH 40         // 32 + 8 pad ushorts -> 80B rows (16B aligned, bank-staggered)
#define MT 128           // batch rows per workgroup

typedef __attribute__((ext_vector_type(16))) __bf16 v16bf;
typedef __attribute__((ext_vector_type(8)))  float  v8f;

__device__ __forceinline__ unsigned short f2bf(float f) {
  unsigned u = __float_as_uint(f);
  u += 0x7FFFu + ((u >> 16) & 1u);          // round-to-nearest-even
  return (unsigned short)(u >> 16);
}
__device__ __forceinline__ float bf2f(unsigned short h) {
  return __uint_as_float(((unsigned)h) << 16);
}

// ---------------- Phase 1: per-class spectral-norm power iteration ----------------
// sigma = u2 . (W v) = ||Wv||^2 / max(||Wv||, eps)   =>  store rsigma = 1/sigma
__global__ void sigma_kernel(const float* __restrict__ W,
                             const float* __restrict__ u,
                             float* __restrict__ rsigma) {
  __shared__ float sU[FF];
  __shared__ float sV[EE];
  __shared__ float red[256];
  const int tid = threadIdx.x;
  const int c = blockIdx.x;
  const float* Wc = W + (size_t)c * FF * EE;

  if (tid < FF) sU[tid] = u[c * FF + tid];
  __syncthreads();

  // v[e] = sum_f W[c,f,e] * u[f] ; thread owns e = tid + 256*j (coalesced)
  float acc[8];
#pragma unroll
  for (int j = 0; j < 8; ++j) acc[j] = 0.f;
  for (int f = 0; f < FF; ++f) {
    const float uf = sU[f];
    const float* row = Wc + (size_t)f * EE + tid;
#pragma unroll
    for (int j = 0; j < 8; ++j) acc[j] = fmaf(row[j * 256], uf, acc[j]);
  }
  float s = 0.f;
#pragma unroll
  for (int j = 0; j < 8; ++j) s += acc[j] * acc[j];
  red[tid] = s;
  __syncthreads();
  for (int k = 128; k > 0; k >>= 1) {
    if (tid < k) red[tid] += red[tid + k];
    __syncthreads();
  }
  const float invv = 1.f / fmaxf(sqrtf(red[0]), 1e-12f);
  __syncthreads();
#pragma unroll
  for (int j = 0; j < 8; ++j) sV[tid + j * 256] = acc[j] * invv;
  __syncthreads();

  // t[f] = sum_e W[c,f,e] * v[e] ; two threads per f, each streams half a row
  const int f = tid >> 1, half = tid & 1;
  const float4* r4 = (const float4*)(Wc + (size_t)f * EE + half * 1024);
  const float4* v4 = (const float4*)(sV + half * 1024);
  float t = 0.f;
  for (int k = 0; k < 256; ++k) {
    float4 a = r4[k], b = v4[k];
    t += a.x * b.x + a.y * b.y + a.z * b.z + a.w * b.w;
  }
  t += __shfl_xor(t, 1, 32);
  red[tid] = (half == 0) ? t * t : 0.f;
  __syncthreads();
  for (int k = 128; k > 0; k >>= 1) {
    if (tid < k) red[tid] += red[tid + k];
    __syncthreads();
  }
  if (tid == 0) {
    const float n2 = red[0];
    const float n = sqrtf(n2);
    rsigma[c] = fmaxf(n, 1e-12f) / n2;   // 1/sigma
  }
}

// ---------------- Phase 2: fused batched GEMM (bf16x3 WMMA) + RBF ----------------
// WG = (class, 128-row m-tile). 8 waves; wave w owns rows [w*16, w*16+16) x all 128 f.
__global__ __launch_bounds__(256) void
fused_kernel(const float* __restrict__ x, const float* __restrict__ W,
             const float* __restrict__ bias, const float* __restrict__ cent,
             const float* __restrict__ rsigma, float* __restrict__ out) {
  __shared__ __align__(16) unsigned short sXH[MT * PITCH];
  __shared__ __align__(16) unsigned short sXL[MT * PITCH];
  __shared__ __align__(16) unsigned short sWH[MT * PITCH];
  __shared__ __align__(16) unsigned short sWL[MT * PITCH];

  const int tid = threadIdx.x;
  const int lane = tid & 31;
  const int wave = tid >> 5;
  const int cls = blockIdx.x;
  const int m0 = blockIdx.y * MT;
  const float* Wc = W + (size_t)cls * FF * EE;

  // staging: 2 threads per row, 16 contiguous floats (one 64B span) each
  const int srow = tid >> 1;
  const int scol = (tid & 1) * 16;
  const float* gx = x + (size_t)(m0 + srow) * EE + scol;
  const float* gw = Wc + (size_t)srow * EE + scol;
  unsigned short* pXH = sXH + srow * PITCH + scol;
  unsigned short* pXL = sXL + srow * PITCH + scol;
  unsigned short* pWH = sWH + srow * PITCH + scol;
  unsigned short* pWL = sWL + srow * PITCH + scol;

  // CDNA5 16-bit A(16x32) frag: lane<16 -> K 0..7 & 16..23, lane>=16 -> K 8..15 & 24..31
  const int aoff = (wave * 16 + (lane & 15)) * PITCH + ((lane >> 4) << 3);
  // CDNA5 16-bit B(32x16) frag: column n = lane&15, lane<16 -> K 0..7 & 16..23, lane>=16 -> K 8..15 & 24..31
  const int boff = (lane & 15) * PITCH + ((lane >> 4) << 3);

  v8f acc[8] = {};

  union Frag { v16bf v; uint4 q[2]; };

  for (int e0 = 0; e0 < EE; e0 += KC) {
    __syncthreads();
#pragma unroll
    for (int q = 0; q < 4; ++q) {
      float4 vx = *(const float4*)(gx + e0 + 4 * q);
      float4 vw = *(const float4*)(gw + e0 + 4 * q);
      unsigned short h0 = f2bf(vx.x), h1 = f2bf(vx.y), h2 = f2bf(vx.z), h3 = f2bf(vx.w);
      uint2 hv; hv.x = (unsigned)h0 | ((unsigned)h1 << 16);
      hv.y = (unsigned)h2 | ((unsigned)h3 << 16);
      *(uint2*)(pXH + 4 * q) = hv;
      unsigned short l0 = f2bf(vx.x - bf2f(h0)), l1 = f2bf(vx.y - bf2f(h1));
      unsigned short l2 = f2bf(vx.z - bf2f(h2)), l3 = f2bf(vx.w - bf2f(h3));
      uint2 lv; lv.x = (unsigned)l0 | ((unsigned)l1 << 16);
      lv.y = (unsigned)l2 | ((unsigned)l3 << 16);
      *(uint2*)(pXL + 4 * q) = lv;

      h0 = f2bf(vw.x); h1 = f2bf(vw.y); h2 = f2bf(vw.z); h3 = f2bf(vw.w);
      hv.x = (unsigned)h0 | ((unsigned)h1 << 16);
      hv.y = (unsigned)h2 | ((unsigned)h3 << 16);
      *(uint2*)(pWH + 4 * q) = hv;
      l0 = f2bf(vw.x - bf2f(h0)); l1 = f2bf(vw.y - bf2f(h1));
      l2 = f2bf(vw.z - bf2f(h2)); l3 = f2bf(vw.w - bf2f(h3));
      lv.x = (unsigned)l0 | ((unsigned)l1 << 16);
      lv.y = (unsigned)l2 | ((unsigned)l3 << 16);
      *(uint2*)(pWL + 4 * q) = lv;
    }
    if (e0 + KC < EE) {   // pull next chunk's cachelines toward L0/L2
      __builtin_prefetch(gx + e0 + KC, 0, 0);
      __builtin_prefetch(gw + e0 + KC, 0, 0);
    }
    __syncthreads();

    Frag aH, aL;
    aH.q[0] = *(const uint4*)(sXH + aoff);
    aH.q[1] = *(const uint4*)(sXH + aoff + 16);
    aL.q[0] = *(const uint4*)(sXL + aoff);
    aL.q[1] = *(const uint4*)(sXL + aoff + 16);
#pragma unroll
    for (int j = 0; j < 8; ++j) {
      const int bo = boff + j * 16 * PITCH;
      Frag bH, bL;
      bH.q[0] = *(const uint4*)(sWH + bo);
      bH.q[1] = *(const uint4*)(sWH + bo + 16);
      bL.q[0] = *(const uint4*)(sWL + bo);
      bL.q[1] = *(const uint4*)(sWL + bo + 16);
      // bf16x3: hi*hi + hi*lo + lo*hi, fp32 accumulate
      acc[j] = __builtin_amdgcn_wmma_f32_16x16x32_bf16(false, aH.v, false, bH.v,
                                                       (short)0, acc[j], false, false);
      acc[j] = __builtin_amdgcn_wmma_f32_16x16x32_bf16(false, aH.v, false, bL.v,
                                                       (short)0, acc[j], false, false);
      acc[j] = __builtin_amdgcn_wmma_f32_16x16x32_bf16(false, aL.v, false, bH.v,
                                                       (short)0, acc[j], false, false);
    }
  }

  // epilogue: z = dot * (1/sigma) + b ; probs = exp(-0.5 * sum_f (c - z)^2)
  const float rsig = rsigma[cls];
  const int n = lane & 15;           // D-matrix column = f within each 16-wide tile
  float bj[8], cj[8];
#pragma unroll
  for (int j = 0; j < 8; ++j) {
    bj[j] = bias[cls * FF + j * 16 + n];
    cj[j] = cent[cls * FF + j * 16 + n];
  }
#pragma unroll
  for (int r = 0; r < 8; ++r) {      // D VGPR r -> row M = r + 8*(lane>=16)
    float s = 0.f;
#pragma unroll
    for (int j = 0; j < 8; ++j) {
      float z = fmaf(acc[j][r], rsig, bj[j]);
      float d = cj[j] - z;
      s = fmaf(d, d, s);
    }
    s += __shfl_xor(s, 1, 32);
    s += __shfl_xor(s, 2, 32);
    s += __shfl_xor(s, 4, 32);
    s += __shfl_xor(s, 8, 32);
    if (n == 0) {
      const int m = m0 + wave * 16 + r + ((lane >> 4) << 3);
      out[(size_t)m * CC + cls] = __expf(-0.5f * s);
    }
  }
}

extern "C" void kernel_launch(void* const* d_in, const int* in_sizes, int n_in,
                              void* d_out, int out_size, void* d_ws, size_t ws_size,
                              hipStream_t stream) {
  (void)in_sizes; (void)n_in; (void)out_size; (void)ws_size;
  const float* x = (const float*)d_in[0];
  const float* W = (const float*)d_in[1];
  const float* b = (const float*)d_in[2];
  const float* u = (const float*)d_in[3];
  const float* c = (const float*)d_in[4];
  float* out = (float*)d_out;
  float* rsig = (float*)d_ws;   // 100 floats of scratch

  sigma_kernel<<<CC, 256, 0, stream>>>(W, u, rsig);
  fused_kernel<<<dim3(CC, BB / MT), 256, 0, stream>>>(x, W, b, c, rsig, out);
}
